// BiologicalRNN_88210038326490
// MI455X (gfx1250) — compile-verified, matrix operand order checked by reading, not run on hardware
//
#include <hip/hip_runtime.h>

typedef __attribute__((ext_vector_type(16))) __bf16 v16bf;
typedef __attribute__((ext_vector_type(8)))  float  v8f;

#define BN 256
#define TN 1000
#define HN 1024
#define ON 16
#define FN 18
#define KTILES 33          // 32 hidden K-tiles + 1 input K-tile
#define NTILES 64          // 1024 / 16
#define GROW 1032          // 1024 + 8 bf16 pad (16B) -> conflict-free strides
#define HROW 516           // 512 + 4 f32 pad

__device__ __forceinline__ unsigned short f2bf(float f) {
  unsigned u = __float_as_uint(f);
  unsigned r = u + 0x7FFFu + ((u >> 16) & 1u);   // round-to-nearest-even
  return (unsigned short)(r >> 16);
}

union FragCast { uint4 q[2]; v16bf v; };

// Pack W_hh (K-tiles 0..31) and masked W_hi (K-tile 32) into WMMA-B fragment
// order: frag(nt,kt): lane l holds 16 bf16 = B[k,n] pairs, n = nt*16+(l&15),
// koff = (v>=4?16:0) + (l>>4)*8 + (v&3)*2 + p.  32 contiguous bytes per lane.
__global__ void pack_w_kernel(const float* __restrict__ W_hh,
                              const float* __restrict__ W_hi,
                              unsigned short* __restrict__ Wp) {
  const int nt   = blockIdx.x / KTILES;
  const int kt   = blockIdx.x % KTILES;
  const int lane = threadIdx.x;
  const int n     = nt * 16 + (lane & 15);
  const int khalf = lane >> 4;
  unsigned short vals[16];
#pragma unroll
  for (int v = 0; v < 8; ++v) {
#pragma unroll
    for (int p = 0; p < 2; ++p) {
      const int koff = ((v >= 4) ? 16 : 0) + khalf * 8 + (v & 3) * 2 + p;
      float val;
      if (kt < 32) {
        val = W_hh[(size_t)n * HN + kt * 32 + koff];        // B[k,n] = W_hh[n,k]
      } else {
        // input K-tile: masked W_hi (input only drives first half of hidden)
        val = (koff < FN && n < HN / 2) ? W_hi[n * FN + koff] : 0.0f;
      }
      vals[v * 2 + p] = f2bf(val);
    }
  }
  uint4* dst = (uint4*)(Wp + (((size_t)nt * KTILES + kt) * 32 + lane) * 16);
  dst[0] = *(uint4*)&vals[0];
  dst[1] = *(uint4*)&vals[8];
}

// Persistent recurrent kernel: 16 blocks x 16 batch rows, 32 waves/block,
// wave w owns N-tiles 2w, 2w+1. Hidden state lives in WMMA accumulators.
__global__ __launch_bounds__(1024, 1) void rnn_kernel(
    const float* __restrict__ x, const float* __restrict__ bias,
    const float* __restrict__ W_oh, const float* __restrict__ hidden_init,
    const unsigned short* __restrict__ Wp,
    float* __restrict__ out, float* __restrict__ hfin) {
  extern __shared__ __align__(16) char smem[];
  unsigned short* gbuf = (unsigned short*)smem;       // elu(h) bf16 [16][GROW]
  unsigned short* xbuf = gbuf + 16 * GROW;            // x_t bf16   [16][32]
  float*          hbuf = (float*)(xbuf + 16 * 32);    // h_new f32  [16][HROW]

  const int tid   = threadIdx.x;
  const int wave  = tid >> 5;
  const int lane  = tid & 31;
  const int lmod  = lane & 15;
  const int lhalf = lane >> 4;
  const int b0    = blockIdx.x * 16;
  const int nt0   = wave * 2;
  const int n0    = nt0 * 16 + lmod;
  const int n1    = n0 + 16;
  const int mC    = lhalf * 8;          // C/D-layout M base for this lane

  const float kk = 0.05f;               // DT/TAU
  const float rr = (1.0f - kk) / kk;    // 19

  float h0[8], h1[8];
  {
    const float hi0 = hidden_init[n0];
    const float hi1 = hidden_init[n1];
#pragma unroll
    for (int v = 0; v < 8; ++v) { h0[v] = hi0; h1[v] = hi1; }
  }
  const float bc0 = bias[n0];
  const float bc1 = bias[n1];

  const uint4* wq = (const uint4*)Wp;
  const uint4* wq0 = wq + ((size_t)(nt0 * KTILES) * 32 + lane) * 2;
  const uint4* wq1 = wq + ((size_t)((nt0 + 1) * KTILES) * 32 + lane) * 2;

  for (int t = 0; t < TN; ++t) {
    // ---- stage x_t (bf16, F padded to 32) ----
    if (tid < 512) {
      const int m = tid >> 5, f = tid & 31;
      const float xv =
          (f < FN) ? x[(size_t)(b0 + m) * TN * FN + (size_t)t * FN + f] : 0.0f;
      xbuf[m * 32 + f] = f2bf(xv);
    }
    // ---- write g = elu(h) to LDS (lane owns M = mC+v, N = n0/n1) ----
#pragma unroll
    for (int v = 0; v < 8; ++v) {
      const int m = mC + v;
      const float a0 = h0[v], a1 = h1[v];
      const float g0 = a0 > 0.0f ? a0 : (__expf(a0) - 1.0f);
      const float g1 = a1 > 0.0f ? a1 : (__expf(a1) - 1.0f);
      gbuf[m * GROW + n0] = f2bf(g0);
      gbuf[m * GROW + n1] = f2bf(g1);
    }
    __syncthreads();

    // ---- seed accumulators: h_new = kk * (19*h + b + W*g + W_hi*x) ----
    v8f c0, c1;
#pragma unroll
    for (int v = 0; v < 8; ++v) {
      c0[v] = h0[v] * rr + bc0;
      c1[v] = h1[v] * rr + bc1;
    }

    const int aoff = lhalf * 8;   // A-layout K sub-offset for this lane
#pragma unroll 4
    for (int kt = 0; kt < 32; ++kt) {
      FragCast af;
      const unsigned short* ap = gbuf + lmod * GROW + kt * 32 + aoff;
      af.q[0] = *(const uint4*)(ap);
      af.q[1] = *(const uint4*)(ap + 16);
      FragCast bf0, bf1;
      const uint4* p0 = wq0 + (size_t)kt * 64;
      const uint4* p1 = wq1 + (size_t)kt * 64;
      bf0.q[0] = p0[0]; bf0.q[1] = p0[1];
      bf1.q[0] = p1[0]; bf1.q[1] = p1[1];
      c0 = __builtin_amdgcn_wmma_f32_16x16x32_bf16(false, af.v, false, bf0.v,
                                                   (short)0, c0, false, false);
      c1 = __builtin_amdgcn_wmma_f32_16x16x32_bf16(false, af.v, false, bf1.v,
                                                   (short)0, c1, false, false);
    }
    {   // input-projection K-tile (kt = 32), A comes from xbuf
      FragCast af;
      const unsigned short* ap = xbuf + lmod * 32 + aoff;
      af.q[0] = *(const uint4*)(ap);
      af.q[1] = *(const uint4*)(ap + 16);
      FragCast bf0, bf1;
      const uint4* p0 = wq0 + (size_t)32 * 64;
      const uint4* p1 = wq1 + (size_t)32 * 64;
      bf0.q[0] = p0[0]; bf0.q[1] = p0[1];
      bf1.q[0] = p1[0]; bf1.q[1] = p1[1];
      c0 = __builtin_amdgcn_wmma_f32_16x16x32_bf16(false, af.v, false, bf0.v,
                                                   (short)0, c0, false, false);
      c1 = __builtin_amdgcn_wmma_f32_16x16x32_bf16(false, af.v, false, bf1.v,
                                                   (short)0, c1, false, false);
    }

#pragma unroll
    for (int v = 0; v < 8; ++v) { h0[v] = kk * c0[v]; h1[v] = kk * c1[v]; }

    // ---- second half of h (output mask) -> LDS for the output projection ----
    if (wave >= 16) {
#pragma unroll
      for (int v = 0; v < 8; ++v) {
        const int m = mC + v;
        hbuf[m * HROW + (n0 - 512)] = h0[v];
        hbuf[m * HROW + (n1 - 512)] = h1[v];
      }
    }
    __syncthreads();

    // ---- out[b,t,o] = sum_{n>=512} h_new[b,n] * W_oh[o,n] ----
    if (tid < 256) {
      const int bl = tid >> 4, o = tid & 15;
      const float* hp = hbuf + bl * HROW;
      const float* wp = W_oh + (size_t)o * HN + 512;
      float acc = 0.0f;
#pragma unroll 8
      for (int j = 0; j < 512; j += 4) {
        const float4 hv = *(const float4*)(hp + j);
        const float4 wv = *(const float4*)(wp + j);
        acc += hv.x * wv.x + hv.y * wv.y + hv.z * wv.z + hv.w * wv.w;
      }
      out[((size_t)(b0 + bl) * TN + t) * ON + o] = acc;
    }
    // next iteration's first __syncthreads() protects gbuf/xbuf/hbuf reuse
  }

  // ---- h_final [B, H] ----
#pragma unroll
  for (int v = 0; v < 8; ++v) {
    const int m = mC + v;
    hfin[(size_t)(b0 + m) * HN + n0] = h0[v];
    hfin[(size_t)(b0 + m) * HN + n1] = h1[v];
  }
}

extern "C" void kernel_launch(void* const* d_in, const int* in_sizes, int n_in,
                              void* d_out, int out_size, void* d_ws,
                              size_t ws_size, hipStream_t stream) {
  const float* x      = (const float*)d_in[0];
  const float* W_hi   = (const float*)d_in[1];
  const float* W_hh   = (const float*)d_in[2];
  const float* bias   = (const float*)d_in[3];
  const float* W_oh   = (const float*)d_in[4];
  const float* hinit  = (const float*)d_in[5];
  (void)in_sizes; (void)n_in; (void)out_size; (void)ws_size;

  unsigned short* Wp = (unsigned short*)d_ws;   // 64*33*512 bf16 = 2.11 MB
  float* out  = (float*)d_out;
  float* hfin = out + (size_t)BN * TN * ON;

  pack_w_kernel<<<NTILES * KTILES, 32, 0, stream>>>(W_hh, W_hi, Wp);

  const size_t smem =
      (size_t)16 * GROW * 2 + (size_t)16 * 32 * 2 + (size_t)16 * HROW * 4;
  rnn_kernel<<<16, 1024, smem, stream>>>(x, bias, W_oh, hinit, Wp, out, hfin);
}
